// LBLResNetBiLm_62689342652480
// MI455X (gfx1250) — matrix operand order repaired and checked by reading, not compiled, not run on hardware
//
#include <hip/hip_runtime.h>
#include <hip/hip_bf16.h>

#define Ll 3
#define Bb 32
#define Ss 512
#define Dd 512
#define Ww 3
#define Mtok (Bb*Ss)   // 16384 tokens per stream

typedef __attribute__((ext_vector_type(16))) __bf16 v16bf;
typedef __attribute__((ext_vector_type(8)))  float  v8f;

__device__ __forceinline__ unsigned short f2bf(float f) {
    union { float f; unsigned u; } v; v.f = f;
    unsigned r = v.u + 0x7FFFu + ((v.u >> 16) & 1u);   // round-to-nearest-even
    return (unsigned short)(r >> 16);
}

// ---------------------------------------------------------------------------
// Weight convert: dst[l][n][k] = bf16(src[l][k][n])   (transposed for B-frag)
// ---------------------------------------------------------------------------
__global__ __launch_bounds__(256) void wconv_kernel(const float* __restrict__ src,
                                                    unsigned short* __restrict__ dst) {
    int idx = blockIdx.x * 256 + threadIdx.x;          // 0 .. L*D*D-1
    int l = idx / (Dd * Dd);
    int r = idx - l * Dd * Dd;
    int n = r >> 9;
    int k = r & 511;
    dst[idx] = f2bf(src[(size_t)l * Dd * Dd + (size_t)k * Dd + n]);
}

// ---------------------------------------------------------------------------
// Windowed weighted sum + LayerNorm. One wave32 per token; lane owns 16 dims.
// ---------------------------------------------------------------------------
__global__ __launch_bounds__(256) void wsum_ln_kernel(
    const float* __restrict__ xbase, int xstride,
    const float* __restrict__ fpad,          // [W, D] front pad (fwd_pad[i])
    const float* __restrict__ bpad,          // [W, D] back pad  (bwd_pad[i])
    const float* __restrict__ w4,            // [W+1] window weights
    const float* __restrict__ g, const float* __restrict__ beta,
    int offset,                              // 0 = fwd, W = bwd
    float* __restrict__ yout,                // [Mtok, D] fp32 (for residual)
    unsigned short* __restrict__ lnout)      // [Mtok, D] bf16 (GEMM A)
{
    int wid  = threadIdx.x >> 5;
    int lane = threadIdx.x & 31;
    int token = blockIdx.x * 8 + wid;
    int b = token >> 9;
    int t = token & 511;

    float w0 = w4[0], w1 = w4[1], w2 = w4[2], w3 = w4[3];

    const float* rp[4];
#pragma unroll
    for (int k = 0; k < 4; ++k) {
        int j = t + offset + k;                        // padded row index
        const float* p;
        if (j < Ww)             p = fpad + (size_t)j * Dd;
        else if (j < Ww + Ss)   p = xbase + (size_t)(b * Ss + (j - Ww)) * xstride;
        else                    p = bpad + (size_t)(j - Ww - Ss) * Dd;
        rp[k] = p;
    }

    int d0 = lane * 16;
    float yv[16];
    float s = 0.f, s2 = 0.f;
#pragma unroll
    for (int u = 0; u < 4; ++u) {
        int d = d0 + u * 4;
        float4 a0 = *(const float4*)(rp[0] + d);
        float4 a1 = *(const float4*)(rp[1] + d);
        float4 a2 = *(const float4*)(rp[2] + d);
        float4 a3 = *(const float4*)(rp[3] + d);
        float4 y;
        y.x = w0 * a0.x + w1 * a1.x + w2 * a2.x + w3 * a3.x;
        y.y = w0 * a0.y + w1 * a1.y + w2 * a2.y + w3 * a3.y;
        y.z = w0 * a0.z + w1 * a1.z + w2 * a2.z + w3 * a3.z;
        y.w = w0 * a0.w + w1 * a1.w + w2 * a2.w + w3 * a3.w;
        yv[u * 4 + 0] = y.x; yv[u * 4 + 1] = y.y;
        yv[u * 4 + 2] = y.z; yv[u * 4 + 3] = y.w;
        s  += y.x + y.y + y.z + y.w;
        s2 += y.x * y.x + y.y * y.y + y.z * y.z + y.w * y.w;
        *(float4*)(yout + (size_t)token * Dd + d) = y;
    }

    // wave32 reduction
#pragma unroll
    for (int m = 16; m >= 1; m >>= 1) {
        s  += __shfl_xor(s,  m, 32);
        s2 += __shfl_xor(s2, m, 32);
    }
    float mean = s * (1.0f / Dd);
    float var  = (s2 - (float)Dd * mean * mean) * (1.0f / (Dd - 1));   // ddof=1
    var = var > 0.f ? var : 0.f;
    float inv = 1.0f / (sqrtf(var) + 1e-6f);

    unsigned int pk[8];
#pragma unroll
    for (int u = 0; u < 8; ++u) {
        int d = d0 + u * 2;
        float l0 = g[d]     * (yv[u * 2]     - mean) * inv + beta[d];
        float l1 = g[d + 1] * (yv[u * 2 + 1] - mean) * inv + beta[d + 1];
        pk[u] = (unsigned)f2bf(l0) | ((unsigned)f2bf(l1) << 16);
    }
    uint4* dst = (uint4*)(lnout + (size_t)token * Dd + d0);
    dst[0] = make_uint4(pk[0], pk[1], pk[2], pk[3]);
    dst[1] = make_uint4(pk[4], pk[5], pk[6], pk[7]);
}

// ---------------------------------------------------------------------------
// bf16 WMMA GEMM, 4x4 fragment blocking: each wave computes a 64x64 tile.
// Block = 8 waves stacked along M (512x64 block tile); the 64-column Bt panel
// (64x512 bf16 = 64 KB, contiguous since Bt is [N,K] row-major) is staged in
// LDS once and shared by all 8 waves.
//   C[m,n] = sum_k A[m,k] * Bt[n,k]
// MODE 0: h = bf16(relu(C + bias))            -> outH
// MODE 1: o = C + bias + resid (fp32, stride) -> outF
// ---------------------------------------------------------------------------
template <int MODE>
__global__ __launch_bounds__(256) void gemm_bf16_kernel(
    const unsigned short* __restrict__ A,    // [Mtok, Dd] bf16 row-major
    const unsigned short* __restrict__ Bt,   // [Dd, Dd] bf16 (transposed weight)
    const float* __restrict__ bias,          // [Dd]
    const float* __restrict__ resid,         // [Mtok, Dd] fp32   (MODE 1)
    float* __restrict__ outF, int outStride, // fp32 out          (MODE 1)
    unsigned short* __restrict__ outH)       // bf16 out [Mtok,Dd](MODE 0)
{
    constexpr int K = Dd;                    // 512
    __shared__ unsigned short sB[64 * Dd];   // 64 KB B panel

    const int tid  = threadIdx.x;
    const int lane = tid & 31;
    const int wid  = tid >> 5;
    const int bM   = blockIdx.x >> 3;        // 0..31  (M / 512)
    const int bN   = blockIdx.x & 7;         // 0..7   (N / 64)
    const int half = lane >> 4;
    const int l15  = lane & 15;

    // ---- stage contiguous 64 KB Bt panel into LDS ----
    {
        const uint4* gq = (const uint4*)(Bt + (size_t)(bN * 64) * K);
        uint4* sq = (uint4*)sB;
#pragma unroll
        for (int i = 0; i < 16; ++i)
            sq[tid + i * 256] = gq[tid + i * 256];    // 4096 uint4 total
    }
    __syncthreads();

    // ---- fragment base pointers ----
    const int wM = bM * 512 + wid * 64;               // wave's M origin
    // A frag (16-bit A 16x32 layout): half 0 -> K [0..7],[16..23]; half 1 -> +8
    const unsigned short* abase = A + (size_t)(wM + l15) * K + half * 8;
    // B frag (same as A): half 0 -> K [0..7],[16..23]; half 1 -> +8
    const unsigned short* bbase = sB + (size_t)l15 * K + half * 8;

    union Frag { v16bf v; uint4 q[2]; };
    v8f acc[16];
#pragma unroll
    for (int i = 0; i < 16; ++i) acc[i] = (v8f){};

#pragma unroll 1
    for (int kk = 0; kk < K; kk += 32) {
        Frag af[4], bf[4];
#pragma unroll
        for (int i = 0; i < 4; ++i) {
            const unsigned short* p = abase + (size_t)i * 16 * K + kk;
            af[i].q[0] = *(const uint4*)(p);
            af[i].q[1] = *(const uint4*)(p + 16);
        }
#pragma unroll
        for (int j = 0; j < 4; ++j) {
            const unsigned short* p = bbase + (size_t)j * 16 * K + kk;
            bf[j].q[0] = *(const uint4*)(p);
            bf[j].q[1] = *(const uint4*)(p + 16);
        }
#pragma unroll
        for (int i = 0; i < 4; ++i)
#pragma unroll
            for (int j = 0; j < 4; ++j)
                acc[i * 4 + j] = __builtin_amdgcn_wmma_f32_16x16x32_bf16(
                    false, af[i].v, false, bf[j].v, (short)0,
                    acc[i * 4 + j], false, false);
    }

    // ---- branch-free epilogue (mode chosen at compile time) ----
#pragma unroll
    for (int i = 0; i < 4; ++i) {
#pragma unroll
        for (int j = 0; j < 4; ++j) {
            const int n = bN * 64 + j * 16 + l15;
            const float bn = bias[n];
            const v8f c = acc[i * 4 + j];
#pragma unroll
            for (int r = 0; r < 8; ++r) {
                const int m = wM + i * 16 + r + half * 8;  // lanes16-31 -> M+8
                float v = c[r] + bn;
                if constexpr (MODE == 0) {
                    v = v > 0.f ? v : 0.f;
                    outH[(size_t)m * Dd + n] = f2bf(v);
                } else {
                    outF[(size_t)m * outStride + n] =
                        v + resid[(size_t)m * Dd + n];
                }
            }
        }
    }
}

// ---------------------------------------------------------------------------
extern "C" void kernel_launch(void* const* d_in, const int* in_sizes, int n_in,
                              void* d_out, int out_size, void* d_ws, size_t ws_size,
                              hipStream_t stream) {
    const float* x        = (const float*)d_in[0];
    const float* fwd_pad  = (const float*)d_in[1];
    const float* bwd_pad  = (const float*)d_in[2];
    const float* fwd_w    = (const float*)d_in[3];
    const float* bwd_w    = (const float*)d_in[4];
    const float* fwd_w1   = (const float*)d_in[5];
    const float* fwd_b1   = (const float*)d_in[6];
    const float* fwd_w2   = (const float*)d_in[7];
    const float* fwd_b2   = (const float*)d_in[8];
    const float* fwd_g    = (const float*)d_in[9];
    const float* fwd_beta = (const float*)d_in[10];
    const float* bwd_w1   = (const float*)d_in[11];
    const float* bwd_b1   = (const float*)d_in[12];
    const float* bwd_w2   = (const float*)d_in[13];
    const float* bwd_b2   = (const float*)d_in[14];
    const float* bwd_g    = (const float*)d_in[15];
    const float* bwd_beta = (const float*)d_in[16];

    char* ws = (char*)d_ws;
    const size_t WSZ = (size_t)Ll * Dd * Dd * 2;       // bf16 weight tensor bytes
    unsigned short* w1f = (unsigned short*)(ws + 0 * WSZ);
    unsigned short* w2f = (unsigned short*)(ws + 1 * WSZ);
    unsigned short* w1b = (unsigned short*)(ws + 2 * WSZ);
    unsigned short* w2b = (unsigned short*)(ws + 3 * WSZ);
    float*          yf32 = (float*)(ws + 4 * WSZ);
    unsigned short* lnb  = (unsigned short*)(ws + 4 * WSZ + (size_t)Mtok * Dd * 4);
    unsigned short* hb   = (unsigned short*)(ws + 4 * WSZ + (size_t)Mtok * Dd * 4
                                                          + (size_t)Mtok * Dd * 2);
    float* out = (float*)d_out;

    // bf16-transpose all four weight stacks once per launch
    const int wblk = (Ll * Dd * Dd) / 256;             // 3072
    wconv_kernel<<<wblk, 256, 0, stream>>>(fwd_w1, w1f);
    wconv_kernel<<<wblk, 256, 0, stream>>>(fwd_w2, w2f);
    wconv_kernel<<<wblk, 256, 0, stream>>>(bwd_w1, w1b);
    wconv_kernel<<<wblk, 256, 0, stream>>>(bwd_w2, w2b);

    const int gemmBlocks = (Mtok / 512) * (Dd / 64);   // 32 * 8 = 256
    for (int i = 0; i < Ll; ++i) {
        for (int s = 0; s < 2; ++s) {
            const float* xb; int xstride;
            if (i == 0) { xb = x; xstride = Dd; }
            else        { xb = out + (size_t)(i - 1) * Mtok * (2 * Dd) + s * Dd;
                          xstride = 2 * Dd; }
            const float* w4 = (s ? bwd_w    : fwd_w)    + i * (Ww + 1);
            const float* g  = (s ? bwd_g    : fwd_g)    + i * Dd;
            const float* bt = (s ? bwd_beta : fwd_beta) + i * Dd;
            wsum_ln_kernel<<<Mtok / 8, 256, 0, stream>>>(
                xb, xstride, fwd_pad + (size_t)i * Ww * Dd,
                bwd_pad + (size_t)i * Ww * Dd, w4, g, bt,
                s ? Ww : 0, yf32, lnb);

            const unsigned short* W1 = (s ? w1b : w1f) + (size_t)i * Dd * Dd;
            const unsigned short* W2 = (s ? w2b : w2f) + (size_t)i * Dd * Dd;
            const float* B1 = (s ? bwd_b1 : fwd_b1) + i * Dd;
            const float* B2 = (s ? bwd_b2 : fwd_b2) + i * Dd;

            // h = relu(ln @ W1 + b1)  -> bf16
            gemm_bf16_kernel<0><<<gemmBlocks, 256, 0, stream>>>(
                lnb, W1, B1, nullptr, nullptr, 0, hb);

            // x' = y + h @ W2 + b2 -> written straight into d_out layer slot
            float* o = out + (size_t)i * Mtok * (2 * Dd) + s * Dd;
            gemm_bf16_kernel<1><<<gemmBlocks, 256, 0, stream>>>(
                hb, W2, B2, yf32, o, 2 * Dd, nullptr);
        }
    }
}
